// Qwen3NextAttention_89678917140579
// MI455X (gfx1250) — matrix operand order repaired and checked by reading, not compiled, not run on hardware
//
#include <hip/hip_runtime.h>
#include <hip/hip_bf16.h>

// ---------------- problem constants ----------------
constexpr int SEQ   = 4096;
constexpr int HID   = 2048;
constexpr int NH    = 16;
constexpr int NKV   = 2;
constexpr int HD    = 256;
constexpr int QKVO  = HD * (NH * 2 + 2 * NKV);   // 9216
constexpr int QSZ2  = 2 * NH * HD;               // 8192 (q+gate region)
constexpr int OUTD  = NH * HD;                   // 4096 (attn flat dim)

// ---------------- wmma types ----------------
typedef __attribute__((ext_vector_type(16))) __bf16          v16bf;
typedef __attribute__((ext_vector_type(16))) unsigned short  u16x16;
typedef __attribute__((ext_vector_type(8)))  float           v8f;

// ---------------- TDM descriptor vector types ----------------
typedef __attribute__((ext_vector_type(4))) unsigned int uint32x4;
typedef __attribute__((ext_vector_type(8))) int          int32x8;
typedef __attribute__((ext_vector_type(4))) int          int32x4;

__device__ inline v16bf as_bf(u16x16 x) {
    union { u16x16 u; v16bf b; } c; c.u = x; return c.b;
}

__device__ inline v8f wmma_bf16(u16x16 a, u16x16 b, v8f c) {
    return __builtin_amdgcn_wmma_f32_16x16x32_bf16(
        false, as_bf(a), false, as_bf(b), (short)0, c, false, false);
}

__device__ inline unsigned short f2bf(float f) {
    unsigned int u = __float_as_uint(f);
    unsigned int r = u + 0x7FFFu + ((u >> 16) & 1u);   // round-to-nearest-even
    return (unsigned short)(r >> 16);
}

union U16x16U { uint4 q[2]; u16x16 v; };

// 16 contiguous bf16 (32 bytes) -> fragment
__device__ inline u16x16 ld16(const unsigned short* p) {
    U16x16U u;
    u.q[0] = *(const uint4*)(p);
    u.q[1] = *(const uint4*)(p + 8);
    return u.v;
}
// two groups of 8 contiguous bf16 -> A-style fragment
__device__ inline u16x16 ld8x2(const unsigned short* p0, const unsigned short* p1) {
    U16x16U u;
    u.q[0] = *(const uint4*)(p0);
    u.q[1] = *(const uint4*)(p1);
    return u.v;
}

// ---------------- TDM: 2D tile load global -> LDS ----------------
// D# per CDNA5 ISA:
//  group0: [1:0]=count(1), [63:32]=lds_addr, [120:64]=global_addr, [127:126]=type(2)
//  group1: [17:16]=data_size(1 -> 2 bytes), [79:48]=tensor_dim0, [111:80]=tensor_dim1,
//          [127:112]=tile_dim0, [143:128]=tile_dim1, [159:144]=tile_dim2,
//          [207:160]=tensor_dim0_stride, [255:208]=tensor_dim1_stride
// This toolchain uses the 6-arg builtin: (g0, g1, g2, g3, g_extra, cpol).
#if __has_builtin(__builtin_amdgcn_tensor_load_to_lds)
#define HAVE_TDM 1
__device__ inline void tdm_load_tile_2d(const unsigned short* gptr, void* lds_ptr,
                                        int dim0, int dim1, int stride0,
                                        int tile0, int tile1) {
    unsigned long long ga = (unsigned long long)(uintptr_t)gptr;
    unsigned int lds_addr = (unsigned int)(uintptr_t)lds_ptr;  // low 32 bits = LDS byte addr
    uint32x4 g0;
    g0[0] = 1u;                                        // count = 1 (valid)
    g0[1] = lds_addr;
    g0[2] = (unsigned int)(ga & 0xffffffffull);
    g0[3] = (unsigned int)((ga >> 32) & 0x01ffffffull) | (2u << 30);  // type=2
    int32x8 g1;
    g1[0] = (int)(1u << 16);                                           // data_size = 2B
    g1[1] = (int)(((unsigned int)dim0 & 0xffffu) << 16);               // dim0[15:0]
    g1[2] = (int)((((unsigned int)dim0 >> 16) & 0xffffu) |
                  (((unsigned int)dim1 & 0xffffu) << 16));             // dim0[31:16] | dim1[15:0]
    g1[3] = (int)((((unsigned int)dim1 >> 16) & 0xffffu) |
                  (((unsigned int)tile0 & 0xffffu) << 16));            // dim1[31:16] | tile0
    g1[4] = (int)((unsigned int)tile1 & 0xffffu);                      // tile1 | tile2(0)
    g1[5] = (int)(unsigned int)stride0;                                // stride0[31:0]
    g1[6] = 0;                                                         // stride0[47:32] | stride1[15:0]
    g1[7] = 0;                                                         // stride1[47:16]
    int32x4 z4 = (int32x4)0;
    int32x8 z8 = (int32x8)0;
    __builtin_amdgcn_tensor_load_to_lds(g0, g1, z4, z4, z8, 0);
}
#else
#define HAVE_TDM 0
#endif

// ---------------- conversion kernels ----------------
__global__ void convert_bf16_kernel(const float* __restrict__ in,
                                    unsigned short* __restrict__ out, size_t n) {
    size_t i = (size_t)blockIdx.x * blockDim.x + threadIdx.x;
    size_t stride = (size_t)gridDim.x * blockDim.x;
    for (; i < n; i += stride) out[i] = f2bf(in[i]);
}

// in: [R][C] f32 row-major -> out: [C][R] bf16 (transposed)
__global__ void transpose_bf16_kernel(const float* __restrict__ in,
                                      unsigned short* __restrict__ out,
                                      int R, int C) {
    size_t n = (size_t)R * C;
    size_t i = (size_t)blockIdx.x * blockDim.x + threadIdx.x;
    size_t stride = (size_t)gridDim.x * blockDim.x;
    for (; i < n; i += stride) {
        int r = (int)(i / C);
        int c = (int)(i - (size_t)r * C);
        out[(size_t)c * R + r] = f2bf(in[i]);
    }
}

// ---------------- tiled bf16 WMMA GEMM (TDM double-buffered) ----------------
// C[M][N] f32 = A[M][K] bf16 (row major) x Bt[N][K] bf16 (B column major)
// block: 256 threads = 8 waves (4 x 2), BM=BN=128, BK=32.
constexpr int BM = 128, BN = 128, BK = 32;

__global__ __launch_bounds__(256) void gemm_bf16_kernel(
        const unsigned short* __restrict__ A,
        const unsigned short* __restrict__ Bt,
        float* __restrict__ C,
        int M, int N, int K) {
    __shared__ unsigned short As[2][BM][BK];   // 2 x 8 KB
    __shared__ unsigned short Bs[2][BN][BK];   // 2 x 8 KB

    const int tid  = threadIdx.x;
    const int lane = tid & 31;
    const int wave = tid >> 5;
    const int wm = wave >> 1;        // 0..3 -> 32-row strip
    const int wn = wave & 1;         // 0..1 -> 64-col strip
    const int lh = lane >> 4;        // half-wave
    const int ll = lane & 15;

    const int bm = blockIdx.y * BM;
    const int bn = blockIdx.x * BN;

    v8f acc[2][4];
#pragma unroll
    for (int i = 0; i < 2; ++i)
#pragma unroll
        for (int j = 0; j < 4; ++j) acc[i][j] = (v8f)0.0f;

    const unsigned short* Abase = A  + (size_t)bm * K;
    const unsigned short* Bbase = Bt + (size_t)bn * K;

#if HAVE_TDM
    // prologue: DMA first tiles into buffer 0 (wave 0 drives the TDM)
    if (wave == 0) {
        tdm_load_tile_2d(Abase, &As[0][0][0], K, M, K, BK, BM);
        tdm_load_tile_2d(Bbase, &Bs[0][0][0], K, N, K, BK, BN);
        __builtin_amdgcn_s_wait_tensorcnt(0);
    }
    __syncthreads();

    int cur = 0;
    for (int k0 = 0; k0 < K; k0 += BK) {
        // kick off next tile DMA into the other buffer while we compute
        if (wave == 0 && (k0 + BK) < K) {
            tdm_load_tile_2d(Abase + k0 + BK, &As[cur ^ 1][0][0], K, M, K, BK, BM);
            tdm_load_tile_2d(Bbase + k0 + BK, &Bs[cur ^ 1][0][0], K, N, K, BK, BN);
        }

        u16x16 af[2], bf[4];
#pragma unroll
        for (int mt = 0; mt < 2; ++mt) {
            int row = wm * 32 + mt * 16 + ll;
            af[mt] = ld8x2(&As[cur][row][lh * 16], &As[cur][row][lh * 16 + 8]);
        }
#pragma unroll
        for (int nt = 0; nt < 4; ++nt) {
            int col = wn * 64 + nt * 16 + ll;
            bf[nt] = ld16(&Bs[cur][col][lh * 16]);
        }
#pragma unroll
        for (int mt = 0; mt < 2; ++mt)
#pragma unroll
            for (int nt = 0; nt < 4; ++nt)
                acc[mt][nt] = wmma_bf16(af[mt], bf[nt], acc[mt][nt]);

        if (wave == 0) __builtin_amdgcn_s_wait_tensorcnt(0);
        __syncthreads();   // next buffer ready + everyone done reading cur
        cur ^= 1;
    }
#else
    // fallback: synchronous staging
    const int srow = tid >> 1;
    const int scol = (tid & 1) * 16;
    const unsigned short* ag = A  + (size_t)(bm + srow) * K + scol;
    const unsigned short* bg = Bt + (size_t)(bn + srow) * K + scol;
    for (int k0 = 0; k0 < K; k0 += BK) {
        uint4 av0 = *(const uint4*)(ag + k0);
        uint4 av1 = *(const uint4*)(ag + k0 + 8);
        uint4 bv0 = *(const uint4*)(bg + k0);
        uint4 bv1 = *(const uint4*)(bg + k0 + 8);
        __syncthreads();
        *(uint4*)&As[0][srow][scol]     = av0;
        *(uint4*)&As[0][srow][scol + 8] = av1;
        *(uint4*)&Bs[0][srow][scol]     = bv0;
        *(uint4*)&Bs[0][srow][scol + 8] = bv1;
        __syncthreads();
        u16x16 af[2], bf[4];
#pragma unroll
        for (int mt = 0; mt < 2; ++mt) {
            int row = wm * 32 + mt * 16 + ll;
            af[mt] = ld8x2(&As[0][row][lh * 16], &As[0][row][lh * 16 + 8]);
        }
#pragma unroll
        for (int nt = 0; nt < 4; ++nt) {
            int col = wn * 64 + nt * 16 + ll;
            bf[nt] = ld16(&Bs[0][col][lh * 16]);
        }
#pragma unroll
        for (int mt = 0; mt < 2; ++mt)
#pragma unroll
            for (int nt = 0; nt < 4; ++nt)
                acc[mt][nt] = wmma_bf16(af[mt], bf[nt], acc[mt][nt]);
    }
#endif

    // epilogue: C/D layout m = r + lh*8, n = ll
#pragma unroll
    for (int mt = 0; mt < 2; ++mt)
#pragma unroll
        for (int nt = 0; nt < 4; ++nt)
#pragma unroll
            for (int r = 0; r < 8; ++r) {
                int row = bm + wm * 32 + mt * 16 + r + lh * 8;
                int col = bn + wn * 64 + nt * 16 + ll;
                C[(size_t)row * N + col] = acc[mt][nt][r];
            }
}

// ---------------- RMSNorm + RoPE + layout kernel ----------------
// qkv: [SEQ][QKVO] f32.  Produces:
//   qb [NH][SEQ][HD] bf16, kb [NKV][SEQ][HD] bf16, vt [NKV][HD][SEQ] bf16
__global__ __launch_bounds__(256) void qkv_post_kernel(
        const float* __restrict__ qkv,
        const int*   __restrict__ positions,
        const float* __restrict__ qw,
        const float* __restrict__ kw,
        unsigned short* __restrict__ qb,
        unsigned short* __restrict__ kb,
        unsigned short* __restrict__ vt) {
    const int t    = blockIdx.x;
    const int wave = threadIdx.x >> 5;
    const int lane = threadIdx.x & 31;
    const float pos = (float)positions[t];
    const float* row = qkv + (size_t)t * QKVO;

    for (int slot = wave; slot < NH + 2 * NKV; slot += 8) {
        const float* src;
        const float* w = nullptr;
        if (slot < NH)            { src = row + slot * (2 * HD);              w = qw; }
        else if (slot < NH + NKV) { src = row + QSZ2 + (slot - NH) * HD;      w = kw; }
        else                      { src = row + QSZ2 + NKV * HD + (slot - NH - NKV) * HD; }

        float v[8];
#pragma unroll
        for (int i = 0; i < 8; ++i) v[i] = src[lane + 32 * i];

        if (w) {
            float ss = 0.0f;
#pragma unroll
            for (int i = 0; i < 8; ++i) ss += v[i] * v[i];
#pragma unroll
            for (int off = 16; off > 0; off >>= 1) ss += __shfl_xor(ss, off, 32);
            float inv = rsqrtf(ss * (1.0f / (float)HD) + 1e-6f);
#pragma unroll
            for (int i = 0; i < 8; ++i)
                v[i] = v[i] * inv * (1.0f + w[lane + 32 * i]);
            // RoPE: freq index j == lane (0..31), pair (d=lane, d=lane+32)
            float freq = pos * __expf(-((float)lane / 32.0f) * 9.210340371976184f);
            float c, s;
            __sincosf(freq, &s, &c);
            float x1 = v[0], x2 = v[1];
            v[0] = x1 * c - x2 * s;
            v[1] = x2 * c + x1 * s;
        }

        if (slot < NH) {
            unsigned short* dst = qb + ((size_t)slot * SEQ + t) * HD;
#pragma unroll
            for (int i = 0; i < 8; ++i) dst[lane + 32 * i] = f2bf(v[i]);
        } else if (slot < NH + NKV) {
            unsigned short* dst = kb + ((size_t)(slot - NH) * SEQ + t) * HD;
#pragma unroll
            for (int i = 0; i < 8; ++i) dst[lane + 32 * i] = f2bf(v[i]);
        } else {
            int kv = slot - NH - NKV;
#pragma unroll
            for (int i = 0; i < 8; ++i)
                vt[((size_t)kv * HD + (lane + 32 * i)) * SEQ + t] = f2bf(v[i]);
        }
    }
}

// ---------------- causal flash attention (WMMA) ----------------
// grid: (NH, SEQ/(16*8)), block 256 = 8 waves, each wave owns one 16-query tile.
__global__ __launch_bounds__(256) void attn_kernel(
        const unsigned short* __restrict__ qb,   // [NH][SEQ][HD]
        const unsigned short* __restrict__ kb,   // [NKV][SEQ][HD]
        const unsigned short* __restrict__ vt,   // [NKV][HD][SEQ]
        const float*          __restrict__ qkv,  // gate source
        unsigned short*       __restrict__ attn) // [SEQ][OUTD]
{
    __shared__ unsigned short p_lds[8][16][32];  // per-wave P scratch (8 KB)

    const int h    = blockIdx.x;
    const int wave = threadIdx.x >> 5;
    const int lane = threadIdx.x & 31;
    const int lh = lane >> 4, ll = lane & 15;
    const int qt = blockIdx.y * 8 + wave;
    const int q0 = qt * 16;
    const int kv = h / (NH / NKV);

    const unsigned short* Qb = qb + ((size_t)h * SEQ + q0) * HD;
    const unsigned short* Kb = kb + (size_t)kv * SEQ * HD;
    const unsigned short* Vb = vt + (size_t)kv * HD * SEQ;

    v8f acc[16];
#pragma unroll
    for (int t2 = 0; t2 < 16; ++t2) acc[t2] = (v8f)0.0f;
    float mrow[8], lrow[8];
#pragma unroll
    for (int r = 0; r < 8; ++r) { mrow[r] = -1e30f; lrow[r] = 0.0f; }

    const float scale = 0.0625f;   // HD^-0.5
    const int kend = q0 + 16;

    for (int kc = 0; kc < kend; kc += 32) {
        // ---- S = Q x K^T over full head dim (two 16-key tiles) ----
        v8f s0 = (v8f)0.0f, s1 = (v8f)0.0f;
#pragma unroll
        for (int dc = 0; dc < HD / 32; ++dc) {
            const int kd = dc * 32;
            u16x16 aq = ld8x2(Qb + (size_t)ll * HD + kd + lh * 16,
                              Qb + (size_t)ll * HD + kd + lh * 16 + 8);
            u16x16 b0 = ld16(Kb + (size_t)(kc + ll) * HD + kd + lh * 16);
            u16x16 b1 = ld16(Kb + (size_t)(kc + 16 + ll) * HD + kd + lh * 16);
            s0 = wmma_bf16(aq, b0, s0);
            s1 = wmma_bf16(aq, b1, s1);
        }

        // ---- mask + online softmax (row r+lh*8 lives in lanes with n=ll) ----
        float chmax[8];
#pragma unroll
        for (int r = 0; r < 8; ++r) {
            int q = q0 + r + lh * 8;
            float a0 = (kc + ll      > q) ? -1e30f : s0[r] * scale;
            float a1 = (kc + 16 + ll > q) ? -1e30f : s1[r] * scale;
            s0[r] = a0; s1[r] = a1;
            float m = fmaxf(a0, a1);
#pragma unroll
            for (int off = 8; off > 0; off >>= 1) m = fmaxf(m, __shfl_xor(m, off, 32));
            chmax[r] = m;
        }
#pragma unroll
        for (int r = 0; r < 8; ++r) {
            float mnew  = fmaxf(mrow[r], chmax[r]);
            float alpha = __expf(mrow[r] - mnew);
            mrow[r] = mnew;
            float p0 = __expf(s0[r] - mnew);
            float p1 = __expf(s1[r] - mnew);
            s0[r] = p0; s1[r] = p1;
            float rs = p0 + p1;
#pragma unroll
            for (int off = 8; off > 0; off >>= 1) rs += __shfl_xor(rs, off, 32);
            lrow[r] = lrow[r] * alpha + rs;
#pragma unroll
            for (int t2 = 0; t2 < 16; ++t2) acc[t2] *= alpha;
        }

        // ---- P: C-layout f32 -> A-layout bf16 via LDS round trip ----
#pragma unroll
        for (int r = 0; r < 8; ++r) {
            int prow = r + lh * 8;
            p_lds[wave][prow][ll]      = f2bf(s0[r]);
            p_lds[wave][prow][16 + ll] = f2bf(s1[r]);
        }
        u16x16 pf = ld8x2(&p_lds[wave][ll][lh * 16],
                          &p_lds[wave][ll][lh * 16 + 8]);

        // ---- O += P x V  (V fragments contiguous thanks to [dim][token]) ----
#pragma unroll
        for (int t2 = 0; t2 < 16; ++t2) {
            int dim = t2 * 16 + ll;
            u16x16 vb = ld16(Vb + (size_t)dim * SEQ + kc + lh * 16);
            acc[t2] = wmma_bf16(pf, vb, acc[t2]);
        }
    }

    // ---- epilogue: normalize, sigmoid gate, store bf16 ----
    float invl[8];
#pragma unroll
    for (int r = 0; r < 8; ++r) invl[r] = 1.0f / lrow[r];
    const float* gb = qkv + (size_t)q0 * QKVO + h * (2 * HD) + HD;
#pragma unroll
    for (int t2 = 0; t2 < 16; ++t2)
#pragma unroll
        for (int r = 0; r < 8; ++r) {
            int m = r + lh * 8;
            int n = t2 * 16 + ll;
            float g   = gb[(size_t)m * QKVO + n];
            float sig = 1.0f / (1.0f + __expf(-g));
            float val = acc[t2][r] * invl[r] * sig;
            attn[(size_t)(q0 + m) * OUTD + h * HD + n] = f2bf(val);
        }
}

// ---------------- launch ----------------
extern "C" void kernel_launch(void* const* d_in, const int* in_sizes, int n_in,
                              void* d_out, int out_size, void* d_ws, size_t ws_size,
                              hipStream_t stream) {
    const float* hidden    = (const float*)d_in[0];
    const int*   positions = (const int*)  d_in[1];
    const float* w_qkv     = (const float*)d_in[2];
    const float* w_o       = (const float*)d_in[3];
    const float* q_norm    = (const float*)d_in[4];
    const float* k_norm    = (const float*)d_in[5];
    float*       out       = (float*)d_out;

    char* ws = (char*)d_ws;
    size_t off = 0;
    auto alloc = [&](size_t bytes) { char* p = ws + off; off += (bytes + 255) & ~(size_t)255; return p; };

    unsigned short* hb    = (unsigned short*)alloc((size_t)SEQ  * HID  * 2);  // hidden bf16
    unsigned short* wqkvt = (unsigned short*)alloc((size_t)QKVO * HID  * 2);  // w_qkv^T bf16
    unsigned short* wot   = (unsigned short*)alloc((size_t)HID  * OUTD * 2);  // w_o^T bf16
    float*          qkvb  = (float*)         alloc((size_t)SEQ  * QKVO * 4);  // qkv f32
    unsigned short* qbuf  = (unsigned short*)alloc((size_t)NH  * SEQ * HD * 2);
    unsigned short* kbuf  = (unsigned short*)alloc((size_t)NKV * SEQ * HD * 2);
    unsigned short* vbuf  = (unsigned short*)alloc((size_t)NKV * HD * SEQ * 2);
    unsigned short* attnb = (unsigned short*)alloc((size_t)SEQ * OUTD * 2);

    // 1. precision conversions / weight transposes
    convert_bf16_kernel<<<4096, 256, 0, stream>>>(hidden, hb, (size_t)SEQ * HID);
    transpose_bf16_kernel<<<8192, 256, 0, stream>>>(w_qkv, wqkvt, HID, QKVO);
    transpose_bf16_kernel<<<8192, 256, 0, stream>>>(w_o, wot, OUTD, HID);

    // 2. QKV projection: [SEQ,HID] x [HID,QKVO] -> f32
    gemm_bf16_kernel<<<dim3(QKVO / BN, SEQ / BM), 256, 0, stream>>>(
        hb, wqkvt, qkvb, SEQ, QKVO, HID);

    // 3. RMSNorm + RoPE + attention layouts
    qkv_post_kernel<<<SEQ, 256, 0, stream>>>(qkvb, positions, q_norm, k_norm,
                                             qbuf, kbuf, vbuf);

    // 4. causal flash attention with sigmoid gate
    attn_kernel<<<dim3(NH, SEQ / 128), 256, 0, stream>>>(qbuf, kbuf, vbuf, qkvb, attnb);

    // 5. output projection: [SEQ,OUTD] x [OUTD,HID] -> d_out f32
    gemm_bf16_kernel<<<dim3(HID / BN, SEQ / BM), 256, 0, stream>>>(
        attnb, wot, out, SEQ, HID, OUTD);
}
